// MultiHeadAttention_609885356075
// MI455X (gfx1250) — matrix operand order repaired and checked by reading, not compiled, not run on hardware
//
#include <hip/hip_runtime.h>
#include <math.h>

#define DIMD 768
#define SEQ  2048
#define NH   8
#define DH   96

typedef __attribute__((ext_vector_type(4)))  float          f32x4;
typedef __attribute__((ext_vector_type(8)))  float          f32x8;
typedef __attribute__((ext_vector_type(4)))  unsigned short u16x4;
typedef __attribute__((ext_vector_type(8)))  unsigned short u16x8;
typedef __attribute__((ext_vector_type(16))) unsigned short u16x16;
typedef __attribute__((ext_vector_type(16))) __bf16         bf16x16;

__device__ inline unsigned short to_bf(float f) {
  unsigned int u = __float_as_uint(f);
  u += 0x7FFFu + ((u >> 16) & 1u);          // round-to-nearest-even
  return (unsigned short)(u >> 16);
}

__device__ inline f32x8 wmma_bf16(u16x16 a, u16x16 b, f32x8 c) {
  return __builtin_amdgcn_wmma_f32_16x16x32_bf16(
      false, __builtin_bit_cast(bf16x16, a),
      false, __builtin_bit_cast(bf16x16, b),
      (short)0, c, false, false);
}

// A-matrix 16x32 fragment (lane = M row in both halves):
//   vgpr j<8 : k = h*8 + j ; vgpr j>=8 : k = 16 + h*8 + (j-8)
__device__ inline u16x16 load_fragA(const unsigned short* p, int h) {
  u16x8 lo = *(const u16x8*)(p + h * 8);
  u16x8 hi = *(const u16x8*)(p + 16 + h * 8);
  u16x16 r;
#pragma unroll
  for (int i = 0; i < 8; ++i) { r[i] = lo[i]; r[i + 8] = hi[i]; }
  return r;
}
// B-matrix 32x16 fragment (lane = N col): j<8 : k = h*8 + j ; j>=8 : k = 16 + h*8 + (j-8)
__device__ inline u16x16 load_fragB(const unsigned short* p, int h) {
  u16x8 lo = *(const u16x8*)(p + h * 8);
  u16x8 hi = *(const u16x8*)(p + 16 + h * 8);
  u16x16 r;
#pragma unroll
  for (int i = 0; i < 8; ++i) { r[i] = lo[i]; r[i + 8] = hi[i]; }
  return r;
}

// ---------------------------------------------------------------------------
// Software-pipelined bf16 WMMA GEMM: C[M,Ncols] = A[M,Kdim] * Bw[Kdim,Ncols]
// EPI==0 : scatter into Q/K/V [B,H,N,dh] bf16 (Q pre-scaled by qscale)
// EPI==1 : C = A*Bw + bias -> f32 Cout
// Block tile 128x128x32, 256 threads = 8 waves (2x4), wave tile 64x32 (4x2).
// Double-buffered LDS; global->reg loads issued before compute of prior tile.
// ---------------------------------------------------------------------------
template <int EPI, typename AT>
__global__ __launch_bounds__(256) void wmma_gemm(
    const AT* __restrict__ A, const float* __restrict__ Bw,
    const float* __restrict__ bias, float* __restrict__ Cout,
    unsigned short* __restrict__ Qb, unsigned short* __restrict__ Kb,
    unsigned short* __restrict__ Vb, int Kdim, int Ncols, float qscale) {
  __shared__ unsigned short Asmem[2][128 * 40];  // [m][k], stride 40 (80B)
  __shared__ unsigned short Bsmem[2][128 * 40];  // [n][k] (transposed from Bw)

  const int tid  = threadIdx.x;
  const int wid  = tid >> 5;
  const int lane = tid & 31;
  const int h    = lane >> 4;
  const int l16  = lane & 15;
  const int m0   = blockIdx.x * 128;
  const int n0   = blockIdx.y * 128;
  const int wm   = (wid >> 2) * 64;
  const int wn   = (wid & 3) * 32;

  const f32x8 z = {0.f, 0.f, 0.f, 0.f, 0.f, 0.f, 0.f, 0.f};
  f32x8 acc[4][2];
#pragma unroll
  for (int fi = 0; fi < 4; ++fi)
#pragma unroll
    for (int fj = 0; fj < 2; ++fj) acc[fi][fj] = z;

  f32x4 aregF[4];
  u16x4 aregH[4];
  float breg[16];

  auto gload = [&](int kk) {             // global -> registers
#pragma unroll
    for (int j = 0; j < 4; ++j) {
      int e4 = j * 256 + tid;            // 1024 quads of A tile 128x32
      int r  = e4 >> 3;
      int c  = (e4 & 7) << 2;
      const AT* src = A + (size_t)(m0 + r) * Kdim + kk + c;
      if constexpr (sizeof(AT) == 4) aregF[j] = *(const f32x4*)src;
      else                           aregH[j] = *(const u16x4*)src;
    }
#pragma unroll
    for (int j = 0; j < 4; ++j) {
      int e4 = j * 256 + tid;            // B tile 32x128, [n][k] target
      int n  = e4 & 127;
      int k4 = (e4 >> 7) << 2;
#pragma unroll
      for (int s = 0; s < 4; ++s)
        breg[j * 4 + s] = Bw[(size_t)(kk + k4 + s) * Ncols + n0 + n];
    }
  };
  auto sstore = [&](int buf) {           // registers -> LDS (bf16)
#pragma unroll
    for (int j = 0; j < 4; ++j) {
      int e4 = j * 256 + tid;
      int r  = e4 >> 3;
      int c  = (e4 & 7) << 2;
      u16x4 q;
      if constexpr (sizeof(AT) == 4) {
#pragma unroll
        for (int s = 0; s < 4; ++s) q[s] = to_bf(aregF[j][s]);
      } else {
        q = aregH[j];
      }
      *(u16x4*)&Asmem[buf][r * 40 + c] = q;
    }
#pragma unroll
    for (int j = 0; j < 4; ++j) {
      int e4 = j * 256 + tid;
      int n  = e4 & 127;
      int k4 = (e4 >> 7) << 2;
      u16x4 q;
#pragma unroll
      for (int s = 0; s < 4; ++s) q[s] = to_bf(breg[j * 4 + s]);
      *(u16x4*)&Bsmem[buf][n * 40 + k4] = q;
    }
  };
  auto compute = [&](int buf) {
    u16x16 afrag[4];
#pragma unroll
    for (int fi = 0; fi < 4; ++fi)
      afrag[fi] = load_fragA(&Asmem[buf][(wm + fi * 16 + l16) * 40], h);
    u16x16 bfrag[2];
#pragma unroll
    for (int fj = 0; fj < 2; ++fj)
      bfrag[fj] = load_fragB(&Bsmem[buf][(wn + fj * 16 + l16) * 40], h);
#pragma unroll
    for (int fj = 0; fj < 2; ++fj)
#pragma unroll
      for (int fi = 0; fi < 4; ++fi)
        acc[fi][fj] = wmma_bf16(afrag[fi], bfrag[fj], acc[fi][fj]);
  };

  const int KS = Kdim >> 5;              // 24 (even)
  gload(0);
  sstore(0);
  __syncthreads();
#pragma unroll 1
  for (int kt = 0; kt < KS; kt += 2) {
    gload((kt + 1) << 5);
    compute(0);
    sstore(1);
    __syncthreads();
    if (kt + 2 < KS) gload((kt + 2) << 5);
    compute(1);
    if (kt + 2 < KS) sstore(0);
    __syncthreads();
  }

  // ---- epilogue ----
#pragma unroll
  for (int fi = 0; fi < 4; ++fi)
#pragma unroll
    for (int fj = 0; fj < 2; ++fj)
#pragma unroll
      for (int i = 0; i < 8; ++i) {
        int gm  = m0 + wm + fi * 16 + h * 8 + i;
        int gn  = n0 + wn + fj * 16 + l16;
        float v = acc[fi][fj][i];
        if constexpr (EPI == 0) {
          int b     = gm >> 11;                 // SEQ = 2048
          int n     = gm & (SEQ - 1);
          int which = gn / DIMD;
          int rem   = gn - which * DIMD;
          int hh    = rem / DH;
          int d     = rem - hh * DH;
          size_t dst = ((size_t)(b * NH + hh) * SEQ + n) * DH + d;
          if (which == 0)      Qb[dst] = to_bf(v * qscale);
          else if (which == 1) Kb[dst] = to_bf(v);
          else                 Vb[dst] = to_bf(v);
        } else {
          Cout[(size_t)gm * Ncols + gn] = v + bias[gn];
        }
      }
}

// ---------------------------------------------------------------------------
// Flash attention per (b,h): 128 threads = 4 waves, wave owns 16 Q rows.
// Q fragments held in registers for the whole kernel; K/V double-buffered in
// LDS with global->reg loads overlapping the previous block's WMMA work.
// ---------------------------------------------------------------------------
__global__ __launch_bounds__(128) void flash_attn(
    const unsigned short* __restrict__ Qb, const unsigned short* __restrict__ Kb,
    const unsigned short* __restrict__ Vb, unsigned short* __restrict__ Ob) {
  __shared__ unsigned short Ks[2][64 * 104];   // [key][d], stride 104 (208B)
  __shared__ unsigned short Vs[2][96 * 72];    // [d][key], stride 72 (144B)
  __shared__ unsigned short Ps[4 * 16 * 72];   // per-wave P 16x64

  const int tid  = threadIdx.x;
  const int wid  = tid >> 5;
  const int lane = tid & 31;
  const int h    = lane >> 4;
  const int l16  = lane & 15;
  const int qblk = blockIdx.x;   // 0..31
  const int bh   = blockIdx.y;   // 0..31
  const size_t base = (size_t)bh * SEQ * DH;

  // Q fragments in registers: 3 dsteps x (16x32) per wave, reused 32 times
  u16x16 aq[3];
  {
    const unsigned short* qrow =
        Qb + base + (size_t)(qblk * 64 + wid * 16 + l16) * DH;
#pragma unroll
    for (int d3 = 0; d3 < 3; ++d3) aq[d3] = load_fragA(qrow + d3 * 32, h);
  }

  const f32x8 z = {0.f, 0.f, 0.f, 0.f, 0.f, 0.f, 0.f, 0.f};
  f32x8 o[6];
  float mrow[8], lrow[8];
#pragma unroll
  for (int d = 0; d < 6; ++d) o[d] = z;
#pragma unroll
  for (int i = 0; i < 8; ++i) { mrow[i] = -1e30f; lrow[i] = 0.f; }

  unsigned short* Pw = &Ps[wid * 16 * 72];

  u16x4 kreg[12], vreg[12];
  auto gload = [&](int kb) {             // global -> registers
    const unsigned short* ksrc = Kb + base + (size_t)kb * 64 * DH;
    const unsigned short* vsrc = Vb + base + (size_t)kb * 64 * DH;
#pragma unroll
    for (int j = 0; j < 12; ++j) {
      int e4 = j * 128 + tid;
      kreg[j] = *(const u16x4*)(ksrc + e4 * 4);
      vreg[j] = *(const u16x4*)(vsrc + e4 * 4);
    }
  };
  auto sstore = [&](int buf) {           // registers -> LDS (V transposed)
#pragma unroll
    for (int j = 0; j < 12; ++j) {
      int e4 = j * 128 + tid;
      int r  = e4 / 24;
      int c  = (e4 % 24) * 4;
      *(u16x4*)&Ks[buf][r * 104 + c] = kreg[j];
#pragma unroll
      for (int s = 0; s < 4; ++s) Vs[buf][(c + s) * 72 + r] = vreg[j][s];
    }
  };
  auto compute = [&](int buf) {
    // S = Q * K^T (scale folded into Q): 4 frags of 16x16 over 64 keys
    f32x8 s[4];
#pragma unroll
    for (int f = 0; f < 4; ++f) s[f] = z;
#pragma unroll
    for (int d3 = 0; d3 < 3; ++d3)
#pragma unroll
      for (int f = 0; f < 4; ++f) {
        u16x16 bk = load_fragB(&Ks[buf][(f * 16 + l16) * 104 + d3 * 32], h);
        s[f] = wmma_bf16(aq[d3], bk, s[f]);
      }
    // online softmax; rows of C live across 16 lanes, row m = h*8+i
#pragma unroll
    for (int i = 0; i < 8; ++i) {
      float mx = fmaxf(fmaxf(s[0][i], s[1][i]), fmaxf(s[2][i], s[3][i]));
#pragma unroll
      for (int off = 8; off; off >>= 1) mx = fmaxf(mx, __shfl_xor(mx, off, 32));
      float mnew  = fmaxf(mrow[i], mx);
      float alpha = __expf(mrow[i] - mnew);
      mrow[i] = mnew;
      float lsum = 0.f;
#pragma unroll
      for (int f = 0; f < 4; ++f) {
        float p = __expf(s[f][i] - mnew);
        lsum += p;
        Pw[(h * 8 + i) * 72 + f * 16 + l16] = to_bf(p);
      }
#pragma unroll
      for (int off = 8; off; off >>= 1) lsum += __shfl_xor(lsum, off, 32);
      lrow[i] = lrow[i] * alpha + lsum;
#pragma unroll
      for (int d = 0; d < 6; ++d) o[d][i] *= alpha;
    }
    asm volatile("s_wait_dscnt 0" ::: "memory");  // wave-local P now readable
    // O += P * V : A = P (16x32 keys), B = V (32 keys x 16 dims)
#pragma unroll
    for (int ks = 0; ks < 2; ++ks) {
      u16x16 ap = load_fragA(&Pw[l16 * 72 + ks * 32], h);
#pragma unroll
      for (int d = 0; d < 6; ++d) {
        u16x16 bv = load_fragB(&Vs[buf][(d * 16 + l16) * 72 + ks * 32], h);
        o[d] = wmma_bf16(ap, bv, o[d]);
      }
    }
  };

  gload(0);
  sstore(0);
  __syncthreads();
#pragma unroll 1
  for (int kb = 0; kb < SEQ / 64; kb += 2) {
    gload(kb + 1);                       // kb+1 <= 31 always valid
    compute(0);
    sstore(1);
    __syncthreads();
    if (kb + 2 < SEQ / 64) gload(kb + 2);
    compute(1);
    if (kb + 2 < SEQ / 64) sstore(0);
    __syncthreads();
  }

  // normalize + write [bh][row][d] bf16 (flat == faithful reshape [B,N,D])
#pragma unroll
  for (int i = 0; i < 8; ++i) {
    float inv = 1.0f / lrow[i];
    int row = qblk * 64 + wid * 16 + h * 8 + i;
    size_t rb = base + (size_t)row * DH;
#pragma unroll
    for (int d = 0; d < 6; ++d)
      Ob[rb + d * 16 + l16] = to_bf(o[d][i] * inv);
  }
}

// ---------------------------------------------------------------------------
extern "C" void kernel_launch(void* const* d_in, const int* in_sizes, int n_in,
                              void* d_out, int out_size, void* d_ws, size_t ws_size,
                              hipStream_t stream) {
  (void)in_sizes; (void)n_in; (void)out_size; (void)ws_size;
  const float* x    = (const float*)d_in[0];   // [4,2048,768]
  const float* Wqkv = (const float*)d_in[1];   // [768,2304]
  const float* Wo   = (const float*)d_in[2];   // [768,768]
  const float* bo   = (const float*)d_in[3];   // [768]
  float* out = (float*)d_out;                  // [4,2048,768]

  const size_t headElems = (size_t)4 * NH * SEQ * DH;  // 6291456
  unsigned short* Qb = (unsigned short*)d_ws;
  unsigned short* Kb = Qb + headElems;
  unsigned short* Vb = Kb + headElems;
  unsigned short* Ob = Vb + headElems;

  const float qscale = 1.0f / sqrtf((float)DH);
  const int M = 4 * SEQ;  // 8192

  dim3 g1(M / 128, (3 * DIMD) / 128);          // 64 x 18
  wmma_gemm<0, float><<<g1, 256, 0, stream>>>(x, Wqkv, nullptr, nullptr,
                                              Qb, Kb, Vb, DIMD, 3 * DIMD, qscale);

  dim3 g2(SEQ / 64, 4 * NH);                   // 32 x 32
  flash_attn<<<g2, 128, 0, stream>>>(Qb, Kb, Vb, Ob);

  dim3 g3(M / 128, DIMD / 128);                // 64 x 6
  wmma_gemm<1, unsigned short><<<g3, 256, 0, stream>>>(Ob, Wo, bo, out,
                                                       nullptr, nullptr, nullptr,
                                                       DIMD, DIMD, 1.0f);
}
